// RecurrentHighwayNetwork_10436770529587
// MI455X (gfx1250) — compile-verified, matrix operand order checked by reading, not run on hardware
//
#include <hip/hip_runtime.h>
#include <hip/hip_bf16.h>
#include <math.h>

// Problem dims (fixed by the reference)
#define Tt 512
#define Bb 64
#define Dd 1024
#define Hh 512
#define Ll 5
#define NBLK_REC 16   // blocks in the persistent recurrent kernel

typedef __attribute__((ext_vector_type(16))) __bf16 v16bf;
typedef __attribute__((ext_vector_type(8)))  float  v8f;

struct U128 { unsigned x, y, z, w; };          // POD 16-byte chunk
struct F4   { float a, b, c, d; };             // POD float4

union BFrag {
  v16bf  v;
  U128   q[2];
  __bf16 s[16];
};

__device__ __forceinline__ float fast_tanh(float x) {
#if __has_builtin(__builtin_amdgcn_tanhf)
  return __builtin_amdgcn_tanhf(x);        // gfx1250 V_TANH_F32 (TRANS32)
#else
  return tanhf(x);
#endif
}

__device__ __forceinline__ float fast_sigmoid(float x) {
#if __has_builtin(__builtin_amdgcn_exp2f)
  const float e = __builtin_amdgcn_exp2f(x * -1.44269504088896f); // exp(-x)
#else
  const float e = __expf(-x);
#endif
  return __builtin_amdgcn_rcpf(1.0f + e);
}

// ---------------------------------------------------------------------------
// K0: zero initial hidden state (both fp32 and bf16 copies) + barrier counter
// ---------------------------------------------------------------------------
__global__ void init_state(float* hf, __bf16* hbf, unsigned* bar) {
  int i = blockIdx.x * blockDim.x + threadIdx.x;
  if (i == 0) *bar = 0u;
  for (int j = i; j < Bb * Hh; j += gridDim.x * blockDim.x) {
    hf[j]  = 0.0f;
    hbf[j] = (__bf16)0.0f;
  }
}

// ---------------------------------------------------------------------------
// K1: pack all weights fp32 -> bf16 in WMMA B-fragment order.
// B-fragment layout for v_wmma_*_16x16x32 (K=32 tile, wave32):
//   lane n%16 holds column n; lanes<16 cover K half {0..7,16..23},
//   lanes>=16 cover {8..15,24..31}; 16 bf16 elems per lane.
// rpack index: ((((l*2+mat)*32 + ct)*16 + kt)*32 + lane)*16 + e
// wpack index: ((((mat   )*32 + ct)*32 + kt)*32 + lane)*16 + e
// ---------------------------------------------------------------------------
__global__ void pack_weights(const float* __restrict__ RH, const float* __restrict__ RC,
                             const float* __restrict__ WH, const float* __restrict__ WC,
                             __bf16* __restrict__ rp, __bf16* __restrict__ wp) {
  const long long NR = (long long)Ll * 2 * Hh * Hh;   // 2,621,440
  const long long NW = (long long)2 * Dd * Hh;        // 1,048,576
  for (long long i = (long long)blockIdx.x * blockDim.x + threadIdx.x;
       i < NR + NW;
       i += (long long)gridDim.x * blockDim.x) {
    if (i < NR) {
      int e    = (int)(i         & 15);
      int lane = (int)((i >> 4)  & 31);
      int kt   = (int)((i >> 9)  & 15);
      int ct   = (int)((i >> 13) & 31);
      int mat  = (int)((i >> 18) & 1);
      int l    = (int)(i >> 19);
      int n = ct * 16 + (lane & 15);
      int k = kt * 32 + ((lane >> 4) << 3) + (e < 8 ? e : e + 8);
      const float* src = mat ? RC : RH;                       // [L,H,H]
      rp[i] = (__bf16)src[((size_t)l * Hh + k) * Hh + n];
    } else {
      long long j = i - NR;
      int e    = (int)(j         & 15);
      int lane = (int)((j >> 4)  & 31);
      int kt   = (int)((j >> 9)  & 31);
      int ct   = (int)((j >> 14) & 31);
      int mat  = (int)(j >> 19);
      int n = ct * 16 + (lane & 15);
      int k = kt * 32 + ((lane >> 4) << 3) + (e < 8 ? e : e + 8);
      const float* src = mat ? WC : WH;                       // [D,H]
      wp[j] = (__bf16)src[(size_t)k * Hh + n];
    }
  }
}

// ---------------------------------------------------------------------------
// K2: XH = x @ W_H + b_H ; XC = x @ W_C + b_C  over all T*B rows.
// One wave per 16x16 output tile; both gates share the A fragment.
// ---------------------------------------------------------------------------
__global__ void __launch_bounds__(256)
input_gemm(const float* __restrict__ x,
           const float* __restrict__ WHb, const float* __restrict__ WCb,
           const __bf16* __restrict__ wp,
           float* __restrict__ XH, float* __restrict__ XC) {
  const int lane   = threadIdx.x & 31;
  const int lane15 = lane & 15;
  const int khalf  = (lane >> 4) << 3;          // 0 or 8
  const int gw = blockIdx.x * (blockDim.x >> 5) + (threadIdx.x >> 5);
  const int nw = gridDim.x * (blockDim.x >> 5);
  const int NT = (Tt * Bb / 16) * (Hh / 16);    // 2048 * 32 = 65536 tiles

  for (int tile = gw; tile < NT; tile += nw) {
    const int rt = tile >> 5;
    const int ct = tile & 31;
    v8f accH = {};
    v8f accC = {};
    const float* xrow = x + (size_t)(rt * 16 + lane15) * Dd;

    for (int kt = 0; kt < 32; ++kt) {
      const int k0 = kt * 32 + khalf;
      F4 p[4];
      p[0] = *(const F4*)(xrow + k0);
      p[1] = *(const F4*)(xrow + k0 + 4);
      p[2] = *(const F4*)(xrow + k0 + 16);
      p[3] = *(const F4*)(xrow + k0 + 20);
      const float* pf = (const float*)p;
      BFrag a;
#pragma unroll
      for (int e = 0; e < 16; ++e) a.s[e] = (__bf16)pf[e];   // v_cvt packed bf16

      const __bf16* bh = wp + (((((size_t)0 * 32 + ct) * 32 + kt) * 32 + lane) << 4);
      const __bf16* bc = wp + (((((size_t)1 * 32 + ct) * 32 + kt) * 32 + lane) << 4);
      BFrag bH, bC;
      bH.q[0] = *(const U128*)bh;        bH.q[1] = *(const U128*)(bh + 8);
      bC.q[0] = *(const U128*)bc;        bC.q[1] = *(const U128*)(bc + 8);

      accH = __builtin_amdgcn_wmma_f32_16x16x32_bf16(false, a.v, false, bH.v,
                                                     (short)0, accH, false, false);
      accC = __builtin_amdgcn_wmma_f32_16x16x32_bf16(false, a.v, false, bC.v,
                                                     (short)0, accC, false, false);
    }

    const int n   = ct * 16 + lane15;
    const float bhb = WHb[n];
    const float bcb = WCb[n];
    const int m0 = rt * 16 + khalf;     // C/D layout: vgpr r -> M = m0 + r
#pragma unroll
    for (int r = 0; r < 8; ++r) {
      XH[(size_t)(m0 + r) * Hh + n] = accH[r] + bhb;
      XC[(size_t)(m0 + r) * Hh + n] = accC[r] + bcb;
    }
  }
}

// ---------------------------------------------------------------------------
// K3: persistent recurrent kernel. 16 blocks * 8 waves = 128 waves,
// one wave per (rowTile rt in 0..3, colTile ct in 0..31) computing BOTH gates.
// 5120 sequential phases separated by an atomic grid barrier.
// ---------------------------------------------------------------------------
__global__ void __launch_bounds__(256)
recurrent(const float* __restrict__ XH, const float* __restrict__ XC,
          const __bf16* __restrict__ rp,
          const float* __restrict__ RHb, const float* __restrict__ RCb,
          float* __restrict__ hf, __bf16* __restrict__ hbf,
          unsigned* bar, float* __restrict__ out) {
  const int lane   = threadIdx.x & 31;
  const int lane15 = lane & 15;
  const int khalf  = (lane >> 4) << 3;                   // 0 or 8
  const int w  = blockIdx.x * (blockDim.x >> 5) + (threadIdx.x >> 5); // 0..127
  const int rt = w >> 5;                                 // 0..3
  const int ct = w & 31;                                 // 0..31
  const int n  = ct * 16 + lane15;

  int phase = 0;
  for (int p = 0; p < 2; ++p) {
    for (int t = 0; t < Tt; ++t) {
      for (int l = 0; l < Ll; ++l) {
        const __bf16* hin_bf = hbf + (size_t)(phase & 1) * (Bb * Hh);
        const float*  hin    = hf  + (size_t)(phase & 1) * (Bb * Hh);
        float*   hout    = hf  + (size_t)((phase & 1) ^ 1) * (Bb * Hh);
        __bf16*  hout_bf = hbf + (size_t)((phase & 1) ^ 1) * (Bb * Hh);

        v8f accH = {};
        v8f accC = {};
        // A-fragment source: bf16 hidden, row = rt*16 + lane15, K pattern via khalf
        const __bf16* arow = hin_bf + (size_t)(rt * 16 + lane15) * Hh + khalf;
        const __bf16* bh0 =
            rp + (((((size_t)(l * 2 + 0) * 32 + ct) * 16) * 32 + lane) << 4);
        const __bf16* bc0 =
            rp + (((((size_t)(l * 2 + 1) * 32 + ct) * 16) * 32 + lane) << 4);

        // Warm next layer's weight stream in L2/L0 (global_prefetch_b8)
        const int ln = (l + 1 < Ll) ? (l + 1) : 0;
        __builtin_prefetch(
            rp + (((((size_t)(ln * 2) * 32 + ct) * 16) * 32 + lane) << 4), 0, 3);

        for (int kt = 0; kt < 16; ++kt) {
          BFrag a, bH, bC;
          a.q[0] = *(const U128*)(arow + kt * 32);
          a.q[1] = *(const U128*)(arow + kt * 32 + 16);
          const __bf16* bh = bh0 + ((size_t)kt << 9); // kt*32*16
          const __bf16* bc = bc0 + ((size_t)kt << 9);
          bH.q[0] = *(const U128*)bh;      bH.q[1] = *(const U128*)(bh + 8);
          bC.q[0] = *(const U128*)bc;      bC.q[1] = *(const U128*)(bc + 8);

          accH = __builtin_amdgcn_wmma_f32_16x16x32_bf16(false, a.v, false, bH.v,
                                                         (short)0, accH, false, false);
          accC = __builtin_amdgcn_wmma_f32_16x16x32_bf16(false, a.v, false, bC.v,
                                                         (short)0, accC, false, false);
        }

        const float pbh = RHb[l * Hh + n];
        const float pbc = RCb[l * Hh + n];
        const int m0 = rt * 16 + khalf;
        const float* xhrow = XH + (size_t)t * Bb * Hh;
        const float* xcrow = XC + (size_t)t * Bb * Hh;

#pragma unroll
        for (int r = 0; r < 8; ++r) {
          const int m = m0 + r;
          float ph = accH[r] + pbh;
          float pc = accC[r] + pbc;
          if (l == 0) {
            ph += xhrow[(size_t)m * Hh + n];
            pc += xcrow[(size_t)m * Hh + n];
          }
          const float hp = hin[(size_t)m * Hh + n];
          const float hl = fast_tanh(ph);
          const float tg = fast_sigmoid(pc);
          const float hn = fmaf(hl - hp, tg, hp);   // hl*t + hp*(1-t)
          hout[(size_t)m * Hh + n]    = hn;
          hout_bf[(size_t)m * Hh + n] = (__bf16)hn;
          if (l == Ll - 1) {
            // outputs[t][m][p*H + n]
            out[((size_t)t * Bb + m) * (2 * Hh) + (size_t)p * Hh + n] = hn;
            if (t == Tt - 1) {
              // hiddens = outputs[-1]
              out[(size_t)Tt * Bb * 2 * Hh + (size_t)m * 2 * Hh +
                  (size_t)p * Hh + n] = hn;
            }
          }
        }

        // ---- grid-wide barrier (all 16 blocks) ----
        ++phase;
        __threadfence();
        __syncthreads();
        if (threadIdx.x == 0) {
          __hip_atomic_fetch_add(bar, 1u, __ATOMIC_RELEASE,
                                 __HIP_MEMORY_SCOPE_AGENT);
          const unsigned target = (unsigned)phase * NBLK_REC;
          while (__hip_atomic_load(bar, __ATOMIC_ACQUIRE,
                                   __HIP_MEMORY_SCOPE_AGENT) < target) {
            __builtin_amdgcn_s_sleep(1);
          }
        }
        __syncthreads();
      }
    }
  }
}

// ---------------------------------------------------------------------------
// Host-side launch
// ---------------------------------------------------------------------------
extern "C" void kernel_launch(void* const* d_in, const int* in_sizes, int n_in,
                              void* d_out, int out_size, void* d_ws, size_t ws_size,
                              hipStream_t stream) {
  const float* x   = (const float*)d_in[0];
  const float* WHw = (const float*)d_in[1];
  const float* WHb = (const float*)d_in[2];
  const float* WCw = (const float*)d_in[3];
  const float* WCb = (const float*)d_in[4];
  const float* RHw = (const float*)d_in[5];
  const float* RHb = (const float*)d_in[6];
  const float* RCw = (const float*)d_in[7];
  const float* RCb = (const float*)d_in[8];

  char* ws = (char*)d_ws;
  size_t off = 0;
  float* XH = (float*)(ws + off);            off += (size_t)Tt * Bb * Hh * 4;  // 64 MB
  float* XC = (float*)(ws + off);            off += (size_t)Tt * Bb * Hh * 4;  // 64 MB
  __bf16* rp = (__bf16*)(ws + off);          off += (size_t)Ll * 2 * Hh * Hh * 2; // 5 MB
  __bf16* wp = (__bf16*)(ws + off);          off += (size_t)2 * Dd * Hh * 2;      // 2 MB
  float* hf = (float*)(ws + off);            off += (size_t)2 * Bb * Hh * 4;
  __bf16* hbf = (__bf16*)(ws + off);         off += (size_t)2 * Bb * Hh * 2;
  unsigned* bar = (unsigned*)(ws + off);     off += 256;

  float* out = (float*)d_out;

  hipLaunchKernelGGL(init_state,   dim3(64),       dim3(256), 0, stream, hf, hbf, bar);
  hipLaunchKernelGGL(pack_weights, dim3(2048),     dim3(256), 0, stream,
                     RHw, RCw, WHw, WCw, rp, wp);
  hipLaunchKernelGGL(input_gemm,   dim3(2048),     dim3(256), 0, stream,
                     x, WHb, WCb, wp, XH, XC);
  hipLaunchKernelGGL(recurrent,    dim3(NBLK_REC), dim3(256), 0, stream,
                     XH, XC, rp, RHb, RCb, hf, hbf, bar, out);
}
